// SingularSelfAttentionHead_48275432407812
// MI455X (gfx1250) — compile-verified, matrix operand order checked by reading, not run on hardware
//
#include <hip/hip_runtime.h>
#include <hip/hip_bf16.h>

typedef __attribute__((ext_vector_type(16))) _Float16 v16h;
typedef __attribute__((ext_vector_type(8)))  _Float16 v8h;
typedef __attribute__((ext_vector_type(4)))  _Float16 v4h;
typedef __attribute__((ext_vector_type(8)))  float    v8f;
typedef __attribute__((ext_vector_type(4)))  int      v4i;

#define B_  8
#define T_  2048
#define D_  1024
#define DH  128   // DQ = DK = DV

#define GAS __attribute__((address_space(1)))
#define LAS __attribute__((address_space(3)))

#if __has_builtin(__builtin_amdgcn_global_load_async_to_lds_b128) && \
    __has_builtin(__builtin_amdgcn_s_wait_asynccnt)
#define HAVE_ASYNC_LDS 1
#else
#define HAVE_ASYNC_LDS 0
#endif

__device__ __forceinline__ void wait_async_lds() {
#if HAVE_ASYNC_LDS
  __builtin_amdgcn_s_wait_asynccnt(0);
#endif
}

// ---------------------------------------------------------------------------
// WMMA f16 16x16x32 fragment loader (ISA 7.12.2 layouts).
// lane L<16: major = L, K chunks [k0,k0+8) and [k0+16,k0+24);
// lane L>=16: major = L-16, K chunks [k0+8,k0+16) and [k0+24,k0+32).
// Works for A (major=row) and B (major=col) as long as K is contiguous.
// ---------------------------------------------------------------------------
__device__ __forceinline__ v16h load_frag_f16(const _Float16* __restrict__ base,
                                              int major0, int ld, int k0) {
  const int lane = threadIdx.x & 31;
  const int l    = lane & 15;
  const int hh   = lane >> 4;
  const _Float16* p = base + (size_t)(major0 + l) * ld + k0 + hh * 8;
  v8h c0 = *(const v8h*)(p);
  v8h c1 = *(const v8h*)(p + 16);
  v16h r;
#pragma unroll
  for (int i = 0; i < 8; ++i) { r[i] = c0[i]; r[i + 8] = c1[i]; }
  return r;
}

__device__ __forceinline__ v8f wmma_f16(v16h a, v16h b, v8f c) {
  return __builtin_amdgcn_wmma_f32_16x16x32_f16(false, a, false, b,
                                                (short)0, c, false, false);
}

// ---------------------------------------------------------------------------
// Prep: fp32 -> fp16 conversion of x (vectorized x4)
// ---------------------------------------------------------------------------
__global__ __launch_bounds__(256) void convert_x_kernel(
    const float* __restrict__ x, _Float16* __restrict__ xh) {
  size_t i = ((size_t)blockIdx.x * blockDim.x + threadIdx.x) * 4;
  float4 v = *(const float4*)(x + i);
  v4h h;
  h[0] = (_Float16)v.x; h[1] = (_Float16)v.y;
  h[2] = (_Float16)v.z; h[3] = (_Float16)v.w;
  *(v4h*)(xh + i) = h;
}

// ---------------------------------------------------------------------------
// Prep: Wt[m][n][d] = W_m[d][n]  (f16, d-contiguous for B-fragment loads)
// ---------------------------------------------------------------------------
__global__ __launch_bounds__(256) void transpose_w_kernel(
    const float* __restrict__ Wq, const float* __restrict__ Wk,
    const float* __restrict__ Wv, _Float16* __restrict__ Wt) {
  int idx = blockIdx.x * 256 + threadIdx.x;      // < 3*DH*D_
  int m   = idx / (DH * D_);
  int rem = idx - m * (DH * D_);
  int n   = rem / D_;
  int d   = rem - n * D_;
  const float* W = (m == 0) ? Wq : ((m == 1) ? Wk : Wv);
  Wt[idx] = (_Float16)W[(size_t)d * DH + n];
}

// ---------------------------------------------------------------------------
// QKV projection.  Block = 8 waves = 128 rows x 128 cols of one matrix.
// W tile (32x128 f16, 8KB) double-buffered in LDS; staged with async
// global->LDS copies (ASYNCcnt) one d-step ahead of the WMMAs.  All 8 W
// fragments are loaded into registers before the 8-WMMA chain so the
// matrix pipe runs back-to-back with staggered ds waits.
// ---------------------------------------------------------------------------
__global__ __launch_bounds__(256) void qkv_proj_kernel(
    const _Float16* __restrict__ xh,   // [B*T][D]
    const _Float16* __restrict__ Wt,   // [3][DH][D]
    const float* __restrict__ bq, const float* __restrict__ bk,
    const float* __restrict__ bv,
    _Float16* __restrict__ Qh,         // [B*T][DH]
    _Float16* __restrict__ Kh,         // [B*T][DH]
    _Float16* __restrict__ Vt)         // [B][DH][T]
{
  __shared__ __align__(16) _Float16 sW[2][32 * DH];   // [buf][n*32 + dk]
  const int wave = threadIdx.x >> 5;
  const int lane = threadIdx.x & 31;
  const int l = lane & 15, hh = lane >> 4;
  const int m    = blockIdx.y;                 // 0=Q 1=K 2=V
  const int row0 = (blockIdx.x * 8 + wave) * 16;
  const _Float16* W = Wt + (size_t)m * DH * D_;

  // stage W[:, d0:d0+32) -> sW[buf] : 512 chunks of 8 halves (16B each)
  auto stage = [&](int buf, int d0) {
    for (int c = threadIdx.x; c < 512; c += 256) {
      int n  = c >> 2;
      int dk = (c & 3) * 8;
      const _Float16* src = &W[(size_t)n * D_ + d0 + dk];
      _Float16*       dst = &sW[buf][n * 32 + dk];
#if HAVE_ASYNC_LDS
      __builtin_amdgcn_global_load_async_to_lds_b128(
          (GAS v4i*)src, (LAS v4i*)dst, 0, 0);
#else
      *(v8h*)dst = *(const v8h*)src;
#endif
    }
  };

  v8f acc[8];
#pragma unroll
  for (int n = 0; n < 8; ++n) acc[n] = (v8f){};

  stage(0, 0);
  v16h a_next = load_frag_f16(xh, row0, D_, 0);
  wait_async_lds();

  for (int step = 0; step < D_ / 32; ++step) {
    __syncthreads();                    // sW[step&1] ready; prev reads done
    v16h a = a_next;
    if (step + 1 < D_ / 32) {           // stage next tile + next A fragment
      stage((step + 1) & 1, (step + 1) * 32);
      a_next = load_frag_f16(xh, row0, D_, (step + 1) * 32);
    }
    const _Float16* wb = &sW[step & 1][0];
    v16h bfr[8];
#pragma unroll
    for (int n = 0; n < 8; ++n) bfr[n] = load_frag_f16(wb, n * 16, 32, 0);
#pragma unroll
    for (int n = 0; n < 8; ++n) acc[n] = wmma_f16(a, bfr[n], acc[n]);
    wait_async_lds();                   // this step's staged copies landed
  }

  const float* bias = (m == 0) ? bq : ((m == 1) ? bk : bv);
  if (m == 2) {
#pragma unroll
    for (int n = 0; n < 8; ++n) {
      float bb = bias[n * 16 + l];
#pragma unroll
      for (int r = 0; r < 8; ++r) {
        int row  = row0 + r + 8 * hh;          // C layout: M = r (+8 hi half)
        int bidx = row >> 11;                  // / T_
        int t    = row & (T_ - 1);
        Vt[((size_t)bidx * DH + n * 16 + l) * T_ + t] =
            (_Float16)(acc[n][r] + bb);
      }
    }
  } else {
    _Float16* dst = (m == 0) ? Qh : Kh;
#pragma unroll
    for (int n = 0; n < 8; ++n) {
      float bb = bias[n * 16 + l];
#pragma unroll
      for (int r = 0; r < 8; ++r) {
        int row = row0 + r + 8 * hh;
        dst[(size_t)row * DH + n * 16 + l] = (_Float16)(acc[n][r] + bb);
      }
    }
  }
}

// ---------------------------------------------------------------------------
// Pass A: colsum[b,k] = sum_{q<=k} exp(S[q,k]).  One key tile per block;
// 8 waves split the triangular q range; Q fragments software-pipelined;
// deterministic LDS reduction.
// ---------------------------------------------------------------------------
__global__ __launch_bounds__(256) void colsum_kernel(
    const _Float16* __restrict__ Qh, const _Float16* __restrict__ Kh,
    float* __restrict__ colsum) {
  __shared__ float sCS[8][16];
  const int wave = threadIdx.x >> 5;
  const int lane = threadIdx.x & 31;
  const int l = lane & 15, hh = lane >> 4;
  const int kt = blockIdx.x;            // key tile (16 columns)
  const int b  = blockIdx.y;
  const int k0 = kt * 16;
  const _Float16* Qb = Qh + (size_t)b * T_ * DH;
  const _Float16* Kb = Kh + (size_t)b * T_ * DH;

  v16h kf[4];
#pragma unroll
  for (int s = 0; s < 4; ++s) kf[s] = load_frag_f16(Kb, k0, DH, s * 32);

  v16h a[4];
  if (wave <= kt) {
#pragma unroll
    for (int s = 0; s < 4; ++s) a[s] = load_frag_f16(Qb, wave * 16, DH, s * 32);
  }

  const int kcol = k0 + l;
  float part = 0.f;
  for (int qt = wave; qt <= kt; qt += 8) {
    v8f s = (v8f){};
#pragma unroll
    for (int st = 0; st < 4; ++st) s = wmma_f16(a[st], kf[st], s);
    if (qt + 8 <= kt) {                 // prefetch next q tile into registers
#pragma unroll
      for (int st = 0; st < 4; ++st)
        a[st] = load_frag_f16(Qb, (qt + 8) * 16, DH, st * 32);
    }
    const int q0 = qt * 16;
#pragma unroll
    for (int r = 0; r < 8; ++r) {
      int q = q0 + r + 8 * hh;
      part += (q > kcol) ? 0.f : __expf(s[r]);
    }
  }
  part += __shfl_xor(part, 16);
  if (lane < 16) sCS[wave][l] = part;
  __syncthreads();
  if (threadIdx.x < 16) {
    float s = 0.f;
#pragma unroll
    for (int w = 0; w < 8; ++w) s += sCS[w][threadIdx.x];
    colsum[(size_t)b * T_ + k0 + threadIdx.x] = s;
  }
}

// ---------------------------------------------------------------------------
// Pass B: out[q,:] = sum_{k>=q} exp(S[q,k])/colsum[k] * V[k,:]
// All 8 K fragments batched before the S-WMMA chain; V fragments hoisted
// above the exp/LDS staging; next-chunk K/V prefetched.  Deterministic
// cross-wave reduction of the 16x128 output in LDS.
// ---------------------------------------------------------------------------
__global__ __launch_bounds__(256) void attn_kernel(
    const _Float16* __restrict__ Qh, const _Float16* __restrict__ Kh,
    const _Float16* __restrict__ Vt, const float* __restrict__ colsum,
    float* __restrict__ out) {
  __shared__ __align__(16) _Float16 sP[8][16 * 32];   // per-wave P staging
  __shared__ float sAll[8][16 * 128];                 // per-wave partial out
  const int wave = threadIdx.x >> 5;
  const int lane = threadIdx.x & 31;
  const int l = lane & 15, hh = lane >> 4;
  const int qt = blockIdx.x;
  const int b  = blockIdx.y;
  const int q0 = qt * 16;
  const _Float16* Qb  = Qh + (size_t)b * T_ * DH;
  const _Float16* Kb  = Kh + (size_t)b * T_ * DH;
  const _Float16* Vb  = Vt + (size_t)b * DH * T_;   // [DH][T], k contiguous
  const float*    csb = colsum + (size_t)b * T_;

  v16h qf[4];
#pragma unroll
  for (int s = 0; s < 4; ++s) qf[s] = load_frag_f16(Qb, q0, DH, s * 32);

  v8f acc[8];
#pragma unroll
  for (int n = 0; n < 8; ++n) acc[n] = (v8f){};

  for (int c = (qt >> 1) + wave; c < T_ / 32; c += 8) {
    const int kb = c * 32;
    if (c + 8 < T_ / 32) {              // prefetch this wave's next chunk
      __builtin_prefetch(&Kb[(size_t)(kb + 256 + lane) * DH], 0, 1);
#pragma unroll
      for (int i = 0; i < 4; ++i)
        __builtin_prefetch(&Vb[(size_t)(i * 32 + lane) * T_ + kb + 256], 0, 1);
    }
    // --- batch all 8 K fragments, then run the 8 S-WMMAs back-to-back ---
    v16h kfr[8];
#pragma unroll
    for (int t2 = 0; t2 < 2; ++t2)
#pragma unroll
      for (int st = 0; st < 4; ++st)
        kfr[t2 * 4 + st] = load_frag_f16(Kb, kb + t2 * 16, DH, st * 32);
    v8f sAcc[2];
    sAcc[0] = (v8f){}; sAcc[1] = (v8f){};
#pragma unroll
    for (int t2 = 0; t2 < 2; ++t2)
#pragma unroll
      for (int st = 0; st < 4; ++st)
        sAcc[t2] = wmma_f16(qf[st], kfr[t2 * 4 + st], sAcc[t2]);
    // --- V fragments issued early: exp VALU work below hides their latency
    v16h vf[8];
#pragma unroll
    for (int n = 0; n < 8; ++n) vf[n] = load_frag_f16(Vb, n * 16, T_, kb);
    // --- mask + exp + colsum scale, staged to per-wave LDS (C -> A layout)
#pragma unroll
    for (int t2 = 0; t2 < 2; ++t2) {
      const int kcol = kb + t2 * 16 + l;
      const float rcs = 1.0f / csb[kcol];
#pragma unroll
      for (int r = 0; r < 8; ++r) {
        int q = q0 + r + 8 * hh;
        float p = (q > kcol) ? 0.f : __expf(sAcc[t2][r]) * rcs;
        sP[wave][(r + 8 * hh) * 32 + t2 * 16 + l] = (_Float16)p;
      }
    }
    // --- P (16x32, A-layout from LDS) @ V ---
    v16h pf = load_frag_f16(&sP[wave][0], 0, 32, 0);
#pragma unroll
    for (int n = 0; n < 8; ++n) acc[n] = wmma_f16(pf, vf[n], acc[n]);
  }

  // deterministic cross-wave reduction
#pragma unroll
  for (int n = 0; n < 8; ++n)
#pragma unroll
    for (int r = 0; r < 8; ++r)
      sAll[wave][(r + 8 * hh) * 128 + n * 16 + l] = acc[n][r];
  __syncthreads();
  float* ob = out + ((size_t)b * T_ + q0) * DH;
  for (int i = threadIdx.x; i < 16 * 128; i += 256) {
    float s = 0.f;
#pragma unroll
    for (int w = 0; w < 8; ++w) s += sAll[w][i];
    ob[i] = s;
  }
}

// ---------------------------------------------------------------------------
extern "C" void kernel_launch(void* const* d_in, const int* in_sizes, int n_in,
                              void* d_out, int out_size, void* d_ws,
                              size_t ws_size, hipStream_t stream) {
  (void)in_sizes; (void)n_in; (void)out_size; (void)ws_size;
  const float* x  = (const float*)d_in[0];
  const float* Wq = (const float*)d_in[1];
  const float* bq = (const float*)d_in[2];
  const float* Wk = (const float*)d_in[3];
  const float* bk = (const float*)d_in[4];
  const float* Wv = (const float*)d_in[5];
  const float* bv = (const float*)d_in[6];
  float* out = (float*)d_out;

  char* ws = (char*)d_ws;
  _Float16* xh = (_Float16*)ws;  ws += (size_t)B_ * T_ * D_ * 2;  // 32 MiB
  _Float16* Wt = (_Float16*)ws;  ws += (size_t)3 * DH * D_ * 2;   // 768 KiB
  _Float16* Qh = (_Float16*)ws;  ws += (size_t)B_ * T_ * DH * 2;  // 4 MiB
  _Float16* Kh = (_Float16*)ws;  ws += (size_t)B_ * T_ * DH * 2;  // 4 MiB
  _Float16* Vt = (_Float16*)ws;  ws += (size_t)B_ * DH * T_ * 2;  // 4 MiB
  float* colsum = (float*)ws;                                     // 64 KiB

  convert_x_kernel<<<(B_ * T_ * D_) / (256 * 4), 256, 0, stream>>>(x, xh);
  transpose_w_kernel<<<(3 * DH * D_) / 256, 256, 0, stream>>>(Wq, Wk, Wv, Wt);
  qkv_proj_kernel<<<dim3(B_ * T_ / 16 / 8, 3), 256, 0, stream>>>(
      xh, Wt, bq, bk, bv, Qh, Kh, Vt);
  colsum_kernel<<<dim3(T_ / 16, B_), 256, 0, stream>>>(Qh, Kh, colsum);
  attn_kernel<<<dim3(T_ / 16, B_), 256, 0, stream>>>(Qh, Kh, Vt, colsum, out);
}
